// DeepGCN_18116172054745
// MI455X (gfx1250) — compile-verified, matrix operand order checked by reading, not run on hardware
//
#include <hip/hip_runtime.h>
#include <hip/hip_bf16.h>
#include <math.h>

// ---------------- CDNA5 types ----------------
typedef __bf16 bf16_t;
typedef __attribute__((ext_vector_type(16))) __bf16 v16bf;
typedef __attribute__((ext_vector_type(8)))  __bf16 v8bf;
typedef __attribute__((ext_vector_type(4)))  __bf16 v4bf;
typedef __attribute__((ext_vector_type(8)))  float  v8f;

// Model constants
#define BB   32
#define NF   192
#define NN   196       // 14*14 tokens
#define NT   (BB*NN)   // 6272 token-rows = 49 * 128 exactly
#define NCLS 1000

__device__ __forceinline__ float gelu_f(float x) {
  return 0.5f * x * (1.0f + erff(x * 0.70710678118654752f));
}

// Build v16bf per CDNA5 16-bit A/B operand layout:
// lane (idx = lane&15, half = lane>>4) holds K runs [8h, 8h+8) in VGPR0-3
// and [16+8h, 16+8h+8) in VGPR4-7. Two 16B loads -> one operand.
__device__ __forceinline__ v16bf combine16(const bf16_t* p0, const bf16_t* p1) {
  v8bf lo = *(const v8bf*)p0;
  v8bf hi = *(const v8bf*)p1;
  v16bf r;
#pragma unroll
  for (int i = 0; i < 8; ++i) { r[i] = lo[i]; r[i + 8] = hi[i]; }
  return r;
}

// ---------------------------------------------------------------------------
// Token-major fused linear:  out[j, o] = epi( sum_c W[o,c] * X[j,c] ),
// j = b*196+n. X bf16 (NT, K), W bf16 (O, K): both WMMA operands are
// contiguous 16B runs in global -> no LDS, no barriers.
// epi: +bias -> (*g + bb) -> optional GELU -> optional +res (f32, token-major)
// Block = 256 thr = 8 waves (2m x 4n); wave tile 32x32 = 4 WMMAs / K-step.
// Outputs: optional f32 master and/or bf16 shadow, both token-major.
// ---------------------------------------------------------------------------
template <int O, int K, int GELU, int RES, int SF32, int SBF16>
__global__ __launch_bounds__(256)
void wmma_linear_tm(const bf16_t* __restrict__ X, const bf16_t* __restrict__ W,
                    const float* __restrict__ bias, const float* __restrict__ g,
                    const float* __restrict__ bb, const float* __restrict__ res,
                    float* __restrict__ outF, bf16_t* __restrict__ outB)
{
  const int tid  = threadIdx.x;
  const int lane = tid & 31, wave = tid >> 5;
  const int half = lane >> 4, l16 = lane & 15;
  const int wm = wave >> 2, wn = wave & 3;          // 2 x 4 wave grid
  const int m0 = blockIdx.y * 64 + wm * 32;
  const int n0 = blockIdx.x * 128 + wn * 32;        // token-row base

  const bf16_t* Wr0 = W + (size_t)(m0 + l16) * K;   // A rows (this lane)
  const bf16_t* Wr1 = Wr0 + (size_t)16 * K;
  const bf16_t* Xr0 = X + (size_t)(n0 + l16) * K;   // B rows (this lane)
  const bf16_t* Xr1 = X + (size_t)(n0 + 16 + l16) * K;

  v8f acc00 = {}, acc01 = {}, acc10 = {}, acc11 = {};
#pragma unroll
  for (int k0 = 0; k0 < K; k0 += 32) {
    const int o0 = k0 + half * 8, o1 = k0 + 16 + half * 8;
    v16bf a0 = combine16(Wr0 + o0, Wr0 + o1);
    v16bf a1 = combine16(Wr1 + o0, Wr1 + o1);
    v16bf b0 = combine16(Xr0 + o0, Xr0 + o1);
    v16bf b1 = combine16(Xr1 + o0, Xr1 + o1);
    acc00 = __builtin_amdgcn_wmma_f32_16x16x32_bf16(false, a0, false, b0, (short)0, acc00, false, false);
    acc01 = __builtin_amdgcn_wmma_f32_16x16x32_bf16(false, a0, false, b1, (short)0, acc01, false, false);
    acc10 = __builtin_amdgcn_wmma_f32_16x16x32_bf16(false, a1, false, b0, (short)0, acc10, false, false);
    acc11 = __builtin_amdgcn_wmma_f32_16x16x32_bf16(false, a1, false, b1, (short)0, acc11, false, false);
  }

  // D layout: col token = l16, row m = r + 8*half -> per lane each acc maps
  // to 8 CONTIGUOUS m values in token-major memory.
#pragma unroll
  for (int sm = 0; sm < 2; ++sm) {
#pragma unroll
    for (int sn = 0; sn < 2; ++sn) {
      v8f acc = sm ? (sn ? acc11 : acc10) : (sn ? acc01 : acc00);
      const int j  = n0 + sn * 16 + l16;
      const int mb = m0 + sm * 16 + half * 8;
      const size_t base = (size_t)j * O + mb;
      float t[8];
#pragma unroll
      for (int r = 0; r < 8; ++r) {
        float v = acc[r] + bias[mb + r];
        v = v * g[mb + r] + bb[mb + r];
        if (GELU) v = gelu_f(v);
        if (RES) v += res[base + r];
        t[r] = v;
      }
      if (SF32) {
        float4* p = (float4*)(outF + base);
        p[0] = make_float4(t[0], t[1], t[2], t[3]);
        p[1] = make_float4(t[4], t[5], t[6], t[7]);
      }
      if (SBF16) {
        v8bf pk;
#pragma unroll
        for (int r = 0; r < 8; ++r) pk[r] = (bf16_t)t[r];
        *(v8bf*)(outB + base) = pk;
      }
    }
  }
}

// ---------------------------------------------------------------------------
// Gram / kNN-key kernel: G[b,q,m] = 2 * <yn_q, yn_m> - sq[m]   (== -dist + cst)
// Yn token-major bf16 (B,196,192). One wave = one 16x16 tile.
// ---------------------------------------------------------------------------
__global__ __launch_bounds__(32)
void wmma_gram_kernel(const bf16_t* __restrict__ Yn, const float* __restrict__ sq,
                      float* __restrict__ G)
{
  const int b  = blockIdx.z;
  const int m0 = blockIdx.y * 16;   // query rows
  const int n0 = blockIdx.x * 16;   // candidate cols
  const int lane = threadIdx.x & 31;
  const int half = lane >> 4, l16 = lane & 15;
  const bf16_t* Yb = Yn + (size_t)b * NN * NF;

  int ra = m0 + l16; if (ra >= NN) ra = NN - 1;   // clamp (padded rows unstored)
  int rb = n0 + l16; if (rb >= NN) rb = NN - 1;

  v8f acc = {};
#pragma unroll
  for (int k0 = 0; k0 < NF; k0 += 32) {
    v16bf a  = combine16(Yb + (size_t)ra * NF + k0 + half * 8,
                         Yb + (size_t)ra * NF + k0 + 16 + half * 8);
    v16bf bm = combine16(Yb + (size_t)rb * NF + k0 + half * 8,
                         Yb + (size_t)rb * NF + k0 + 16 + half * 8);
    acc = __builtin_amdgcn_wmma_f32_16x16x32_bf16(false, a, false, bm,
                                                  (short)0, acc, false, false);
  }
  const float* sqb = sq + (size_t)b * NN;
#pragma unroll
  for (int r = 0; r < 8; ++r) {
    int q = m0 + r + 8 * half;
    int m = n0 + l16;
    if (q < NN && m < NN)
      G[((size_t)b * NN + q) * NN + m] = 2.f * acc[r] - sqb[m];
  }
}

// ---------------- scalar kernels ----------------

// Weight f32 -> bf16 (vectorized x4)
__global__ void cvt_w_kernel(const float* __restrict__ w, bf16_t* __restrict__ o, int n4)
{
  int t = blockIdx.x * blockDim.x + threadIdx.x;
  if (t >= n4) return;
  float4 v = ((const float4*)w)[t];
  v4bf pk; pk[0] = (bf16_t)v.x; pk[1] = (bf16_t)v.y; pk[2] = (bf16_t)v.z; pk[3] = (bf16_t)v.w;
  ((v4bf*)o)[t] = pk;
}

// Direct 3x3 conv NCHW, pad 1, fused bias -> BN -> optional GELU
__global__ void conv3x3_kernel(const float* __restrict__ x, const float* __restrict__ w,
                               const float* __restrict__ bias, const float* __restrict__ g,
                               const float* __restrict__ bb, float* __restrict__ out,
                               int Ci, int Co, int Hi, int Wi, int Ho, int Wo,
                               int stride, int gelu_flag)
{
  long long t = (long long)blockIdx.x * blockDim.x + threadIdx.x;
  long long total = (long long)BB * Co * Ho * Wo;
  if (t >= total) return;
  int ox = (int)(t % Wo);
  int oy = (int)((t / Wo) % Ho);
  int oc = (int)((t / ((long long)Wo * Ho)) % Co);
  int b  = (int)(t / ((long long)Wo * Ho * Co));

  float acc = bias[oc];
  const float* xb = x + (size_t)b * Ci * Hi * Wi;
  const float* wk = w + (size_t)oc * Ci * 9;
  for (int ic = 0; ic < Ci; ++ic) {
    const float* xc = xb + (size_t)ic * Hi * Wi;
    const float* wc = wk + ic * 9;
#pragma unroll
    for (int ky = 0; ky < 3; ++ky) {
      int iy = oy * stride + ky - 1;
      if ((unsigned)iy >= (unsigned)Hi) continue;
#pragma unroll
      for (int kx = 0; kx < 3; ++kx) {
        int ix = ox * stride + kx - 1;
        if ((unsigned)ix >= (unsigned)Wi) continue;
        acc += xc[(size_t)iy * Wi + ix] * wc[ky * 3 + kx];
      }
    }
  }
  acc = acc * g[oc] + bb[oc];
  if (gelu_flag) acc = gelu_f(acc);
  out[t] = acc;
}

// NCHW conv5 output + pos_embed -> token-major f32 master + bf16 shadow
__global__ void pack_h_kernel(const float* __restrict__ hin, const float* __restrict__ pos,
                              float* __restrict__ hf, bf16_t* __restrict__ hb)
{
  long long t = (long long)blockIdx.x * blockDim.x + threadIdx.x;
  if (t >= (long long)BB * NF * NN) return;
  int n = (int)(t % NN);
  int c = (int)((t / NN) % NF);
  int b = (int)(t / ((long long)NN * NF));
  float v = hin[t] + pos[(size_t)c * NN + n];
  size_t o = ((size_t)b * NN + n) * NF + c;
  hf[o] = v; hb[o] = (bf16_t)v;
}

// L2-normalize each token (token-major): yn bf16 + squared norm
__global__ void norm_tm_kernel(const float* __restrict__ y, bf16_t* __restrict__ yn,
                               float* __restrict__ sq)
{
  int t = blockIdx.x * blockDim.x + threadIdx.x;   // j = b*NN + n
  if (t >= NT) return;
  const float* p = y + (size_t)t * NF;
  float s = 0.f;
  for (int c = 0; c < NF; ++c) s += p[c] * p[c];
  float inv = 1.f / fmaxf(sqrtf(s), 1e-12f);
  bf16_t* o = yn + (size_t)t * NF;
  float s2 = 0.f;
  for (int c = 0; c < NF; ++c) {
    float v = p[c] * inv;
    o[c] = (bf16_t)v;
    s2 += v * v;
  }
  sq[t] = s2;
}

// Top-(k*d) by key (== -dist ordering, first-index tie-break), keep every d-th
__global__ void topk_kernel(const float* __restrict__ G, int* __restrict__ idxo,
                            int k, int d)
{
  int row = blockIdx.x * blockDim.x + threadIdx.x;   // b*NN + q
  if (row >= NT) return;
  const float* v = G + (size_t)row * NN;
  unsigned long long used[4] = {0ull, 0ull, 0ull, 0ull};
  int kd = k * d;
  for (int p = 0; p < kd; ++p) {
    float best = -3.4e38f; int bi = 0;
    for (int m = 0; m < NN; ++m) {
      if ((used[m >> 6] >> (m & 63)) & 1ull) continue;
      float val = v[m];
      if (val > best) { best = val; bi = m; }
    }
    used[bi >> 6] |= 1ull << (bi & 63);
    if (p % d == 0) idxo[(size_t)row * k + p / d] = bi;
  }
}

// Max-relative gather (token-major, c fastest -> coalesced gathers);
// z interleaved: z[j, 2c]=y, z[j, 2c+1]=max_i(y[idx_i]-y), bf16 out
__global__ void mrz_tm_kernel(const float* __restrict__ y, const int* __restrict__ idx,
                              bf16_t* __restrict__ z, int k)
{
  long long t = (long long)blockIdx.x * blockDim.x + threadIdx.x;
  if (t >= (long long)NT * NF) return;
  int c = (int)(t % NF);
  int j = (int)(t / NF);                 // b*NN + n
  int bbase = (j / NN) * NN;             // batch row offset for gathers
  float yv = y[(size_t)j * NF + c];
  const int* id = idx + (size_t)j * k;
  float mr = -3.4e38f;
  for (int i = 0; i < k; ++i)
    mr = fmaxf(mr, y[(size_t)(bbase + id[i]) * NF + c] - yv);
  bf16_t* zr = z + (size_t)j * (2 * NF);
  zr[2 * c]     = (bf16_t)yv;
  zr[2 * c + 1] = (bf16_t)mr;
}

__global__ void pool_tm_kernel(const float* __restrict__ h, float* __restrict__ pooled)
{
  int t = blockIdx.x * blockDim.x + threadIdx.x;   // b*NF + o
  if (t >= BB * NF) return;
  int b = t / NF, o = t - b * NF;
  const float* p = h + (size_t)b * NN * NF + o;
  float s = 0.f;
  for (int n = 0; n < NN; ++n) s += p[(size_t)n * NF];
  pooled[t] = s * (1.f / NN);
}

__global__ void pred1_kernel(const float* __restrict__ pooled, const float* __restrict__ w,
                             const float* __restrict__ bias, const float* __restrict__ g,
                             const float* __restrict__ bb, float* __restrict__ z1)
{
  int t = blockIdx.x * blockDim.x + threadIdx.x;   // b*1024 + i
  if (t >= BB * 1024) return;
  int b = t >> 10, i = t & 1023;
  const float* pv = pooled + (size_t)b * NF;
  const float* wr = w + (size_t)i * NF;
  float s = bias[i];
  for (int c = 0; c < NF; ++c) s += pv[c] * wr[c];
  z1[t] = gelu_f(s * g[i] + bb[i]);
}

__global__ void pred2_kernel(const float* __restrict__ z1, const float* __restrict__ w,
                             const float* __restrict__ bias, float* __restrict__ out)
{
  int t = blockIdx.x * blockDim.x + threadIdx.x;   // b*NCLS + j
  if (t >= BB * NCLS) return;
  int b = t / NCLS, j = t - b * NCLS;
  const float* zv = z1 + (size_t)b * 1024;
  const float* wr = w + (size_t)j * 1024;
  float s = bias[j];
  for (int c = 0; c < 1024; ++c) s += zv[c] * wr[c];
  out[t] = s;
}

// ---------------- workspace layout (bytes, 256B-aligned) ----------------
// stem chain (x -> s1 -> s2 -> s3 -> s4 -> c5(=s1) -> h/hb):
#define OFF_S1 0ull           // 32*24*112*112 f32 (also conv5 NCHW out later)
#define OFF_S2 38535168ull    // 32*48*56*56 f32
#define OFF_S3 57802752ull    // 32*96*28*28 f32
#define OFF_S4 67436544ull    // 32*192*14*14 f32
#define OFF_H  72253440ull    // (NT,192) f32 token-major residual master
#define OFF_HB 77070336ull    // (NT,192) bf16 shadow       -> end 79478784
// block-phase temps alias dead s1/s2 region (all < OFF_S4):
#define OFF_Y    0ull          // (NT,192) f32 token-major
#define OFF_YN   4816896ull    // (NT,192) bf16
#define OFF_SQ   7225344ull    // (NT,)    f32
#define OFF_GRAM 7250432ull    // (B,196,196) f32
#define OFF_IDX  12167680ull   // (NT,18)  int
#define OFF_Z    12619264ull   // (NT,384) bf16
#define OFF_Z2   17436160ull   // (NT,384) bf16
#define OFF_T    22253056ull   // (NT,768) bf16
#define OFF_WSC  31886848ull   // weight scratch bf16 (max 147456 elems)
#define OFF_POOL 32181760ull   // (32,192) f32
#define OFF_Z1   32206336ull   // (32,1024) f32   -> end 32337408
// total ws requirement ~79.5 MB

static inline int cdiv(long long a, int b) { return (int)((a + b - 1) / b); }

extern "C" void kernel_launch(void* const* d_in, const int* in_sizes, int n_in,
                              void* d_out, int out_size, void* d_ws, size_t ws_size,
                              hipStream_t stream)
{
  (void)in_sizes; (void)n_in; (void)out_size; (void)ws_size;
  // d_in order = setup_inputs insertion order:
  // [0]=x, [1..20]=stem (w,b,g,bb)x5, [21]=pos_embed,
  // [22..261]=12 blocks x 20 (fc1 w/b/g/bb, gc..., fc2..., ffn1..., ffn2...),
  // [262..267]=pred w1,b1,g1,bb1,w2,b2
  const float* x   = (const float*)d_in[0];
  const float* pos = (const float*)d_in[21];
  char* ws = (char*)d_ws;
  float*  s1   = (float*)(ws + OFF_S1);
  float*  s2   = (float*)(ws + OFF_S2);
  float*  s3   = (float*)(ws + OFF_S3);
  float*  s4   = (float*)(ws + OFF_S4);
  float*  hf   = (float*)(ws + OFF_H);
  bf16_t* hb   = (bf16_t*)(ws + OFF_HB);
  float*  y    = (float*)(ws + OFF_Y);
  bf16_t* yn   = (bf16_t*)(ws + OFF_YN);
  float*  sq   = (float*)(ws + OFF_SQ);
  float*  gram = (float*)(ws + OFF_GRAM);
  int*    idx  = (int*)(ws + OFF_IDX);
  bf16_t* z    = (bf16_t*)(ws + OFF_Z);
  bf16_t* z2   = (bf16_t*)(ws + OFF_Z2);
  bf16_t* tbuf = (bf16_t*)(ws + OFF_T);
  bf16_t* wsc  = (bf16_t*)(ws + OFF_WSC);
  float*  pool = (float*)(ws + OFF_POOL);
  float*  z1   = (float*)(ws + OFF_Z1);

  auto cvt = [&](const float* w, int nelem) {
    cvt_w_kernel<<<cdiv(nelem / 4, 256), 256, 0, stream>>>(w, wsc, nelem / 4);
  };

  // ---- stem: 5 fused conv+BN(+GELU), then token-major pack + pos ----
  const int  ch[6] = {3, 24, 48, 96, 192, 192};
  const int  hw[6] = {224, 112, 56, 28, 14, 14};
  float* sbuf[6] = { (float*)x, s1, s2, s3, s4, s1 /* conv5 reuses dead s1 */ };
  for (int i = 0; i < 5; ++i) {
    long long total = (long long)BB * ch[i + 1] * hw[i + 1] * hw[i + 1];
    conv3x3_kernel<<<cdiv(total, 256), 256, 0, stream>>>(
        sbuf[i], (const float*)d_in[1 + i * 4], (const float*)d_in[2 + i * 4],
        (const float*)d_in[3 + i * 4], (const float*)d_in[4 + i * 4], sbuf[i + 1],
        ch[i], ch[i + 1], hw[i], hw[i], hw[i + 1], hw[i + 1],
        (i < 4) ? 2 : 1, (i < 4) ? 1 : 0);
  }
  pack_h_kernel<<<cdiv((long long)NT * NF, 256), 256, 0, stream>>>(s1, pos, hf, hb);

  // ---- 12 grapher + FFN blocks ----
  const int knn[12] = {9, 9, 10, 11, 12, 13, 13, 14, 15, 16, 17, 18};
  const int dil[12] = {1, 1, 1, 1, 2, 2, 2, 2, 3, 3, 3, 3};

  for (int i = 0; i < 12; ++i) {
    const float* const* p = (const float* const*)&d_in[22 + i * 20];
    // fc1: y(f32) = BN(W1 h + b1)
    cvt(p[0], NF * NF);
    wmma_linear_tm<NF, NF, 0, 0, 1, 0><<<dim3(49, NF / 64), 256, 0, stream>>>(
        hb, wsc, p[1], p[2], p[3], nullptr, y, nullptr);
    // normalize tokens -> bf16 + |.|^2
    norm_tm_kernel<<<cdiv(NT, 256), 256, 0, stream>>>(y, yn, sq);
    // kNN keys via WMMA Gram
    wmma_gram_kernel<<<dim3(13, 13, BB), 32, 0, stream>>>(yn, sq, gram);
    // top-(k*d), dilated
    topk_kernel<<<cdiv(NT, 64), 64, 0, stream>>>(gram, idx, knn[i], dil[i]);
    // max-relative + interleaved stack -> z bf16 (NT,384)
    mrz_tm_kernel<<<cdiv((long long)NT * NF, 256), 256, 0, stream>>>(y, idx, z, knn[i]);
    // gc: z2(bf16) = GELU(BN(Wg z + bg))
    cvt(p[4], 2 * NF * 2 * NF);
    wmma_linear_tm<2 * NF, 2 * NF, 1, 0, 0, 1><<<dim3(49, (2 * NF) / 64), 256, 0, stream>>>(
        z, wsc, p[5], p[6], p[7], nullptr, nullptr, z2);
    // fc2: h = BN(W2 z2 + b2) + h   (f32 master + bf16 shadow)
    cvt(p[8], NF * 2 * NF);
    wmma_linear_tm<NF, 2 * NF, 0, 1, 1, 1><<<dim3(49, NF / 64), 256, 0, stream>>>(
        z2, wsc, p[9], p[10], p[11], hf, hf, hb);
    // ffn1: t(bf16) = GELU(BN(Wf1 h + bf1))
    cvt(p[12], 4 * NF * NF);
    wmma_linear_tm<4 * NF, NF, 1, 0, 0, 1><<<dim3(49, (4 * NF) / 64), 256, 0, stream>>>(
        hb, wsc, p[13], p[14], p[15], nullptr, nullptr, tbuf);
    // ffn2: h = BN(Wf2 t + bf2) + h
    cvt(p[16], NF * 4 * NF);
    wmma_linear_tm<NF, 4 * NF, 0, 1, 1, 1><<<dim3(49, NF / 64), 256, 0, stream>>>(
        tbuf, wsc, p[17], p[18], p[19], hf, hf, hb);
  }

  // ---- head ----
  pool_tm_kernel<<<cdiv(BB * NF, 256), 256, 0, stream>>>(hf, pool);
  pred1_kernel<<<cdiv(BB * 1024, 256), 256, 0, stream>>>(
      pool, (const float*)d_in[262], (const float*)d_in[263],
      (const float*)d_in[264], (const float*)d_in[265], z1);
  pred2_kernel<<<cdiv(BB * NCLS, 256), 256, 0, stream>>>(
      z1, (const float*)d_in[266], (const float*)d_in[267], (float*)d_out);
}
